// MessagePassingModel_79431125172597
// MI455X (gfx1250) — matrix operand order repaired and checked by reading, not compiled, not run on hardware
//
#include <hip/hip_runtime.h>
#include <hip/hip_bf16.h>

// ---------------------------------------------------------------------------
// MessagePassingModel on MI455X (gfx1250, wave32, WMMA).
// All GEMMs use v_wmma_f32_16x16x32_f16 (K=32 matches the radial basis size).
// All WMMA operands are PRE-SWIZZLED into the per-lane VGPR layout so each
// lane fetches its 16 f16 operand values with one aligned 32-byte load
// (global_load_b128 x2 / ds_load_b128 x2) instead of 16 scalar u16 loads.
// Segment-sum exploits the all-pairs edge layout (63 consecutive edges per
// dst atom) so no global atomics are needed.
// ---------------------------------------------------------------------------

typedef __attribute__((ext_vector_type(16))) _Float16 v16h;
typedef __attribute__((ext_vector_type(8)))  _Float16 v8h;
typedef __attribute__((ext_vector_type(8)))  float    v8f;

#define B_   16
#define N_   64
#define A_   (B_ * N_)          // 1024 atoms
#define F_   64
#define K_   32
#define EPB  (N_ * (N_ - 1))    // 4032 edges per batch
#define EPA  (N_ - 1)           // 63 edges per dst atom
#define E_   (B_ * EPB)         // 64512 edges total
#define NITER 2
#define CUTOFF_ 5.0f
#define COULOMB_SCALE_ 7.199822675975274f
#define INV_SQRT3_ 0.57735026918962576451f

__constant__ float c_binom[K_] = {
    1.f, 31.f, 465.f, 4495.f, 31465.f, 169911.f, 736281.f, 2629575.f,
    7888725.f, 20160075.f, 44352165.f, 84672315.f, 141120525.f, 206253075.f,
    265182525.f, 300540195.f, 300540195.f, 265182525.f, 206253075.f,
    141120525.f, 84672315.f, 44352165.f, 20160075.f, 7888725.f, 2629575.f,
    736281.f, 169911.f, 31465.f, 4495.f, 465.f, 31.f, 1.f};

__device__ __forceinline__ float sigmoidf_(float x) {
    return 1.0f / (1.0f + __expf(-x));
}

__device__ __forceinline__ v8f wmma16x16x32(v16h a, v16h b, v8f c) {
    return __builtin_amdgcn_wmma_f32_16x16x32_f16(
        /*neg_a=*/false, a, /*neg_b=*/false, b,
        /*c_mod=*/(short)0, c, /*reuse_a=*/false, /*reuse_b=*/false);
}

// Swizzled B-tile fetch: one aligned 32B load per lane.
__device__ __forceinline__ v16h b_tile_sw(const _Float16* Wsw, int kblk, int t,
                                          int lane) {
    return *(const v16h*)(Wsw + (((size_t)(kblk * 4 + t)) * 32 + (size_t)lane) * 16);
}

// A-tile from f32 memory (global or LDS) in the CDNA5 16-bit A layout:
// lane<16 : row=lane,    K = k0 + {0..7, 16..23}
// lane>=16: row=lane-16, K = k0 + {8..15, 24..31}
__device__ __forceinline__ v16h a_tile_f32(const float* base, int rowStride,
                                           int k0, int lane) {
    int r  = lane & 15;
    int hl = lane >> 4;
    const float* row = base + r * rowStride + k0 + 8 * hl;
    float4 lo0 = *(const float4*)(row + 0);
    float4 lo1 = *(const float4*)(row + 4);
    float4 hi0 = *(const float4*)(row + 16);
    float4 hi1 = *(const float4*)(row + 20);
    v16h a;
    a[0]  = (_Float16)lo0.x; a[1]  = (_Float16)lo0.y;
    a[2]  = (_Float16)lo0.z; a[3]  = (_Float16)lo0.w;
    a[4]  = (_Float16)lo1.x; a[5]  = (_Float16)lo1.y;
    a[6]  = (_Float16)lo1.z; a[7]  = (_Float16)lo1.w;
    a[8]  = (_Float16)hi0.x; a[9]  = (_Float16)hi0.y;
    a[10] = (_Float16)hi0.z; a[11] = (_Float16)hi0.w;
    a[12] = (_Float16)hi1.x; a[13] = (_Float16)hi1.y;
    a[14] = (_Float16)hi1.z; a[15] = (_Float16)hi1.w;
    return a;
}

// ---------------------------------------------------------------------------
// Kernel 1: weight fp32->fp16 SWIZZLED conversion, ef projection, x init.
// Swizzled layout per matrix: [(kblk*4 + ntile)*32 + lane]*16 + p, with
//   n = ntile*16 + (lane&15), k = kblk*32 + 8*(lane>>4) + (p&7) + 16*(p>>3).
// ---------------------------------------------------------------------------
__global__ void prep_kernel(const float* __restrict__ mp_w,
                            const float* __restrict__ dense_w,
                            const float* __restrict__ td0,
                            const float* __restrict__ td1,
                            const float* __restrict__ Ef,
                            const float* __restrict__ ef_kernel,
                            const float* __restrict__ ef_bias,
                            const int*   __restrict__ Z,
                            const float* __restrict__ embed,
                            _Float16* __restrict__ wh,
                            float* __restrict__ efp,
                            float* __restrict__ xbuf) {
    const int stride = gridDim.x * blockDim.x;
    const int tid    = blockIdx.x * blockDim.x + threadIdx.x;
    const int n_mp = NITER * 4 * K_ * F_;   // 16384 (8 mats x 2048)
    const int n_dw = NITER * 3 * F_ * F_;   // 24576 (6 mats x 4096)
    const int n_td = F_ * F_;               // 4096

    // mp weights: K=32 -> 1 kblk, 4 ntiles per matrix
    for (int idx = tid; idx < n_mp; idx += stride) {
        int mat = idx >> 11, rem = idx & 2047;
        int t = rem >> 9, rem2 = rem & 511;
        int lane = rem2 >> 4, p = rem2 & 15;
        int n = t * 16 + (lane & 15);
        int k = 8 * (lane >> 4) + (p & 7) + 16 * (p >> 3);
        wh[idx] = (_Float16)mp_w[mat * (K_ * F_) + k * F_ + n];
    }
    // dense / td weights: K=64 -> 2 kblks, 4 ntiles per matrix
    for (int idx = tid; idx < n_dw + 2 * n_td; idx += stride) {
        int mat = idx >> 12, rem = idx & 4095;
        int kb = rem >> 11, rem2 = rem & 2047;
        int t = rem2 >> 9, rem3 = rem2 & 511;
        int lane = rem3 >> 4, p = rem3 & 15;
        int n = t * 16 + (lane & 15);
        int k = kb * 32 + 8 * (lane >> 4) + (p & 7) + 16 * (p >> 3);
        const float* src;
        if (mat < NITER * 3) src = dense_w + mat * (F_ * F_);
        else if (mat == NITER * 3) src = td0;
        else src = td1;
        wh[n_mp + idx] = (_Float16)src[k * F_ + n];
    }
    for (int idx = tid; idx < B_ * F_; idx += stride) {
        int bb = idx >> 6, f = idx & 63;
        efp[idx] = Ef[bb * 3 + 0] * ef_kernel[0 * F_ + f] +
                   Ef[bb * 3 + 1] * ef_kernel[1 * F_ + f] +
                   Ef[bb * 3 + 2] * ef_kernel[2 * F_ + f] + ef_bias[f];
    }
    for (int idx = tid; idx < A_ * 4 * F_; idx += stride) {
        int aa = idx >> 8, rm = idx & 255, m = rm >> 6, f = rm & 63;
        xbuf[idx] = (m == 0) ? embed[Z[aa] * F_ + f] : 0.0f;
    }
}

// ---------------------------------------------------------------------------
// Kernel 2: per-edge geometry + Bernstein radial basis, written directly in
// the WMMA A-matrix lane layout, padded to 64 edge-slots per dst atom.
// Slot j (tile w=j/16, row r=j%16):
//   lane r    gets K {0..7,16..23}, lane r+16 gets K {8..15,24..31}.
// ---------------------------------------------------------------------------
__device__ __forceinline__ v8h pack8(const float* v, int k0, float scale) {
    v8h h;
#pragma unroll
    for (int p = 0; p < 8; ++p) h[p] = (_Float16)(v[k0 + p] * scale);
    return h;
}

__global__ void geom_kernel(const float* __restrict__ pos,
                            const int* __restrict__ src_idx,
                            _Float16* __restrict__ b0sw,
                            _Float16* __restrict__ b1sw,
                            float* __restrict__ rbuf) {
    int g = blockIdx.x * blockDim.x + threadIdx.x;   // slot index
    if (g >= A_ * 64) return;
    int a = g >> 6, j = g & 63;
    int b = a >> 6, i = a & 63;
    int w = j >> 4, r = j & 15;

    size_t base0 = ((size_t)(a * 4 + w) * 32) * 16;
    _Float16* d0 = b0sw + base0;

    if (j == EPA) {   // padded slot: zero-fill A-tile rows
        v8h z = {};
        *(v8h*)(d0 + (size_t)r * 16)            = z;
        *(v8h*)(d0 + (size_t)r * 16 + 8)        = z;
        *(v8h*)(d0 + (size_t)(r + 16) * 16)     = z;
        *(v8h*)(d0 + (size_t)(r + 16) * 16 + 8) = z;
#pragma unroll
        for (int m = 0; m < 3; ++m) {
            _Float16* d1 = b1sw + (((size_t)(a * 3 + m) * 4 + w) * 32) * 16;
            *(v8h*)(d1 + (size_t)r * 16)            = z;
            *(v8h*)(d1 + (size_t)r * 16 + 8)        = z;
            *(v8h*)(d1 + (size_t)(r + 16) * 16)     = z;
            *(v8h*)(d1 + (size_t)(r + 16) * 16 + 8) = z;
        }
        return;
    }

    int sl = src_idx[i * EPA + j];
    int d  = a;                    // dst atom (consecutive-edge layout)
    int s  = sl + b * N_;
    float dx = pos[s * 3 + 0] - pos[d * 3 + 0];
    float dy = pos[s * 3 + 1] - pos[d * 3 + 1];
    float dz = pos[s * 3 + 2] - pos[d * 3 + 2];
    float rr = sqrtf(dx * dx + dy * dy + dz * dz);
    rbuf[b * EPB + i * EPA + j] = rr;
    float rs = fmaxf(rr, 1e-8f);
    float ux = dx / rs, uy = dy / rs, uz = dz / rs;
    float t   = 1.0f / (1.0f + rr);
    float omt = 1.0f - t;
    float u   = (rr / CUTOFF_); u = u * u;
    float cut = (u < 1.0f) ? __expf(1.0f - 1.0f / fmaxf(1.0f - u, 1e-12f)) : 0.0f;

    float tp[K_], b0v[K_];
    float v = 1.0f;
#pragma unroll
    for (int k = 0; k < K_; ++k) { tp[k] = v; v *= t; }
    float ov = 1.0f;
#pragma unroll
    for (int k = K_ - 1; k >= 0; --k) { b0v[k] = c_binom[k] * tp[k] * ov * cut; ov *= omt; }

    *(v8h*)(d0 + (size_t)r * 16)            = pack8(b0v, 0, 1.0f);
    *(v8h*)(d0 + (size_t)r * 16 + 8)        = pack8(b0v, 16, 1.0f);
    *(v8h*)(d0 + (size_t)(r + 16) * 16)     = pack8(b0v, 8, 1.0f);
    *(v8h*)(d0 + (size_t)(r + 16) * 16 + 8) = pack8(b0v, 24, 1.0f);
    const float sc[3] = {ux, uy, uz};
#pragma unroll
    for (int m = 0; m < 3; ++m) {
        _Float16* d1 = b1sw + (((size_t)(a * 3 + m) * 4 + w) * 32) * 16;
        *(v8h*)(d1 + (size_t)r * 16)            = pack8(b0v, 0, sc[m]);
        *(v8h*)(d1 + (size_t)r * 16 + 8)        = pack8(b0v, 16, sc[m]);
        *(v8h*)(d1 + (size_t)(r + 16) * 16)     = pack8(b0v, 8, sc[m]);
        *(v8h*)(d1 + (size_t)(r + 16) * 16 + 8) = pack8(b0v, 24, sc[m]);
    }
}

// ---------------------------------------------------------------------------
// Kernel 3: message passing.  One block (4 waves) per dst atom; each wave owns
// a 16-edge M-tile.  A/B operands are single aligned v16h loads.  Elementwise
// combine with gathered x[src] on the D-tile layout, ds_add_f32 reduction.
// ---------------------------------------------------------------------------
__global__ __launch_bounds__(128) void mp_kernel(const float* __restrict__ x,
                                                 float* __restrict__ y,
                                                 const _Float16* __restrict__ b0sw,
                                                 const _Float16* __restrict__ b1sw,
                                                 const int* __restrict__ src_idx,
                                                 const _Float16* __restrict__ mpw,
                                                 int lastIter) {
    __shared__ __align__(32) _Float16 sW[4 * K_ * F_];  // w00,w11,w01,w10 swizzled
    __shared__ int   sSrc[EPA + 1];
    __shared__ float sY[4 * F_];

    const int tid = threadIdx.x, lane = tid & 31, w = tid >> 5;
    const int a = blockIdx.x, b = a >> 6, i = a & 63;

    // stage this iteration's 4 swizzled weight matrices (16 KB) into LDS
    const uint4* wsrc = (const uint4*)mpw;
    uint4* wdst = (uint4*)sW;
    for (int idx = tid; idx < 4 * K_ * F_ / 8; idx += 128) wdst[idx] = wsrc[idx];
    for (int idx = tid; idx < 4 * F_; idx += 128) sY[idx] = 0.0f;
    if (tid < EPA) sSrc[tid] = src_idx[i * EPA + tid] + b * N_;
    if (tid == EPA) sSrc[EPA] = a;   // dummy for padded slot 63
    __syncthreads();

    v16h a0  = *(const v16h*)(b0sw + ((size_t)(a * 4 + w) * 32 + lane) * 16);
    v16h a1x = *(const v16h*)(b1sw + (((size_t)(a * 3 + 0) * 4 + w) * 32 + lane) * 16);
    v16h a1y = *(const v16h*)(b1sw + (((size_t)(a * 3 + 1) * 4 + w) * 32 + lane) * 16);
    v16h a1z = *(const v16h*)(b1sw + (((size_t)(a * 3 + 2) * 4 + w) * 32 + lane) * 16);

    const _Float16* w00 = sW;
    const _Float16* w11 = sW + K_ * F_;
    const _Float16* w01 = sW + 2 * K_ * F_;
    const _Float16* w10 = sW + 3 * K_ * F_;

    const int hl = lane >> 4, nl = lane & 15;

#pragma unroll
    for (int t = 0; t < 4; ++t) {
        v8f c00 = {}, c11x = {}, c11y = {}, c11z = {};
        c00  = wmma16x16x32(a0,  b_tile_sw(w00, 0, t, lane), c00);
        c11x = wmma16x16x32(a1x, b_tile_sw(w11, 0, t, lane), c11x);
        c11y = wmma16x16x32(a1y, b_tile_sw(w11, 0, t, lane), c11y);
        c11z = wmma16x16x32(a1z, b_tile_sw(w11, 0, t, lane), c11z);
        v8f c01x = {}, c01y = {}, c01z = {}, c10 = {};
        if (!lastIter) {   // grid-uniform branch: EXEC stays all-ones
            c01x = wmma16x16x32(a1x, b_tile_sw(w01, 0, t, lane), c01x);
            c01y = wmma16x16x32(a1y, b_tile_sw(w01, 0, t, lane), c01y);
            c01z = wmma16x16x32(a1z, b_tile_sw(w01, 0, t, lane), c01z);
            c10  = wmma16x16x32(a0,  b_tile_sw(w10, 0, t, lane), c10);
        }
        const int n = 16 * t + nl;
        float acc0 = 0.f, acc1 = 0.f, acc2 = 0.f, acc3 = 0.f;
#pragma unroll
        for (int v = 0; v < 8; ++v) {
            int j  = 16 * w + 8 * hl + v;      // edge slot in this atom's list
            int sa = sSrc[j];                  // j==63 -> dummy (c's are 0)
            float xs0 = x[sa * 256 + 0 * F_ + n];
            float xs1 = x[sa * 256 + 1 * F_ + n];
            float xs2 = x[sa * 256 + 2 * F_ + n];
            float xs3 = x[sa * 256 + 3 * F_ + n];
            acc0 += xs0 * c00[v] +
                    (xs1 * c11x[v] + xs2 * c11y[v] + xs3 * c11z[v]) * INV_SQRT3_;
            if (!lastIter) {
                acc1 += xs0 * c01x[v] + xs1 * c10[v];
                acc2 += xs0 * c01y[v] + xs2 * c10[v];
                acc3 += xs0 * c01z[v] + xs3 * c10[v];
            }
        }
        atomicAdd(&sY[0 * F_ + n], acc0);      // ds_add_f32
        if (!lastIter) {
            atomicAdd(&sY[1 * F_ + n], acc1);
            atomicAdd(&sY[2 * F_ + n], acc2);
            atomicAdd(&sY[3 * F_ + n], acc3);
        }
    }
    __syncthreads();
    for (int idx = tid; idx < 4 * F_; idx += 128) y[a * 256 + idx] = sY[idx];
}

// ---------------------------------------------------------------------------
// Kernel 4: per-atom update (gated-SiLU + 3 dense layers), 4 atoms per wave.
// Row r = localAtom*4 + channel, so a 16x64 WMMA tile covers 4 atoms.
// ---------------------------------------------------------------------------
__device__ __forceinline__ void gsilu_lds(float* T, float* G, int lane) {
    for (int idx = lane; idx < 4 * F_; idx += 32) {
        int la = idx >> 6, f = idx & 63;
        G[idx] = sigmoidf_(T[(la * 4) * F_ + f]);
    }
    __syncthreads();
    for (int idx = lane; idx < 16 * F_; idx += 32) {
        int r = idx >> 6;
        T[idx] *= G[(r >> 2) * F_ + (idx & 63)];
    }
    __syncthreads();
}

__device__ __forceinline__ void dense_lds(const float* In, float* Out,
                                          const _Float16* Wsw,
                                          const float* bias, int lane) {
    const int hl = lane >> 4, nl = lane & 15;
    v16h a0 = a_tile_f32(In, F_, 0, lane);
    v16h a1 = a_tile_f32(In, F_, 32, lane);
#pragma unroll
    for (int t = 0; t < 4; ++t) {
        v8f c = {};
        c = wmma16x16x32(a0, b_tile_sw(Wsw, 0, t, lane), c);
        c = wmma16x16x32(a1, b_tile_sw(Wsw, 1, t, lane), c);
        const int n = 16 * t + nl;
#pragma unroll
        for (int v = 0; v < 8; ++v) {
            int r = 8 * hl + v;
            Out[r * F_ + n] = c[v] + (((r & 3) == 0) ? bias[n] : 0.0f);
        }
    }
    __syncthreads();
}

__global__ __launch_bounds__(32) void update_kernel(float* __restrict__ x,
                                                    const float* __restrict__ y,
                                                    const float* __restrict__ efp,
                                                    const _Float16* __restrict__ Wh,
                                                    const float* __restrict__ bias) {
    __shared__ __align__(16) float T[16 * F_], U[16 * F_], V[16 * F_], G[4 * F_];
    const int lane  = threadIdx.x;
    const int aBase = blockIdx.x * 4;
    const int batch = aBase >> 6;

    for (int idx = lane; idx < 16 * F_; idx += 32)
        T[idx] = x[aBase * 256 + idx] + y[aBase * 256 + idx];
    __syncthreads();
    gsilu_lds(T, G, lane);                       // x = gsilu(x + y)

    dense_lds(T, U, Wh + 0 * F_ * F_, bias + 0 * F_, lane);
    gsilu_lds(U, G, lane);                       // x = gsilu(edense0(x))

    for (int idx = lane; idx < 16 * F_; idx += 32)
        V[idx] = U[idx] + efp[batch * F_ + (idx & 63)];
    __syncthreads();
    dense_lds(V, T, Wh + 1 * F_ * F_, bias + 1 * F_, lane);  // xEF
    for (int idx = lane; idx < 16 * F_; idx += 32) U[idx] += T[idx];
    __syncthreads();                             // x = x + xEF

    dense_lds(U, T, Wh + 2 * F_ * F_, bias + 2 * F_, lane);
    gsilu_lds(T, G, lane);                       // x = gsilu(edense2(x))

    for (int idx = lane; idx < 16 * F_; idx += 32)
        x[aBase * 256 + idx] = T[idx];
}

// ---------------------------------------------------------------------------
// Kernel 5: output heads.  16 atoms per wave; t0/t1 via WMMA, F->1 via VALU.
// ---------------------------------------------------------------------------
__global__ __launch_bounds__(32) void heads_kernel(
    const float* __restrict__ x, const _Float16* __restrict__ td0sw,
    const _Float16* __restrict__ td1sw, const float* __restrict__ dip_w,
    const float* __restrict__ charge_w, const float* __restrict__ charge_b,
    const float* __restrict__ energy_w, const float* __restrict__ energy_b,
    const float* __restrict__ element_bias, const int* __restrict__ Z,
    float* __restrict__ qbuf, float* __restrict__ aen,
    float* __restrict__ adip) {
    __shared__ __align__(16) float T0[16 * F_];
    __shared__ __align__(16) float T1[3][16 * F_];
    const int lane  = threadIdx.x;
    const int aBase = blockIdx.x * 16;
    const int hl = lane >> 4, nl = lane & 15;

    {   // t0 = x0 @ td_w0
        v16h a0 = a_tile_f32(x + (size_t)aBase * 256, 256, 0, lane);
        v16h a1 = a_tile_f32(x + (size_t)aBase * 256, 256, 32, lane);
#pragma unroll
        for (int t = 0; t < 4; ++t) {
            v8f c = {};
            c = wmma16x16x32(a0, b_tile_sw(td0sw, 0, t, lane), c);
            c = wmma16x16x32(a1, b_tile_sw(td0sw, 1, t, lane), c);
            int n = 16 * t + nl;
#pragma unroll
            for (int v = 0; v < 8; ++v) T0[(8 * hl + v) * F_ + n] = c[v];
        }
    }
#pragma unroll
    for (int m = 0; m < 3; ++m) {    // t1 = x[:,1:4,:] @ td_w1
        const float* base = x + (size_t)aBase * 256 + (m + 1) * F_;
        v16h a0 = a_tile_f32(base, 256, 0, lane);
        v16h a1 = a_tile_f32(base, 256, 32, lane);
#pragma unroll
        for (int t = 0; t < 4; ++t) {
            v8f c = {};
            c = wmma16x16x32(a0, b_tile_sw(td1sw, 0, t, lane), c);
            c = wmma16x16x32(a1, b_tile_sw(td1sw, 1, t, lane), c);
            int n = 16 * t + nl;
#pragma unroll
            for (int v = 0; v < 8; ++v) T1[m][(8 * hl + v) * F_ + n] = c[v];
        }
    }
    __syncthreads();
    for (int idx = lane; idx < 16 * F_; idx += 32) {   // gate by sigmoid(t0)
        float g = sigmoidf_(T0[idx]);
        T0[idx] *= g; T1[0][idx] *= g; T1[1][idx] *= g; T1[2][idx] *= g;
    }
    __syncthreads();
    if (lane < 16) {
        int atom = aBase + lane;
        float d0 = 0.f;
        for (int f = 0; f < F_; ++f) d0 += T0[lane * F_ + f] * dip_w[f];
        float sd0 = sigmoidf_(d0);
        for (int m = 0; m < 3; ++m) {
            float d1 = 0.f;
            for (int f = 0; f < F_; ++f) d1 += T1[m][lane * F_ + f] * dip_w[f];
            adip[atom * 3 + m] = d1 * sd0;
        }
        float qc = 0.f, en = 0.f;
        for (int f = 0; f < F_; ++f) {
            float x0 = x[(size_t)atom * 256 + f];
            qc += x0 * charge_w[f];
            en += x0 * energy_w[f];
        }
        qc += charge_b[0];
        qbuf[atom] = qc * sigmoidf_(qc);               // silu
        aen[atom]  = en + energy_b[0] + element_bias[Z[atom]];
    }
}

// ---------------------------------------------------------------------------
// Kernel 6: per-batch reduction: energy (+ Coulomb) and dipole -> d_out.
// ---------------------------------------------------------------------------
__global__ __launch_bounds__(64) void finalize_kernel(
    const float* __restrict__ pos, const int* __restrict__ dst_idx,
    const int* __restrict__ src_idx, const float* __restrict__ rbuf,
    const float* __restrict__ qbuf, const float* __restrict__ aen,
    const float* __restrict__ adip, float* __restrict__ out) {
    __shared__ float s[8];   // 0..2 pos sums, 3 energy, 4..6 dipole, 7 coulomb
    const int b = blockIdx.x, t = threadIdx.x;
    if (t < 8) s[t] = 0.0f;
    __syncthreads();
    const int atom = b * N_ + t;
    float px = pos[atom * 3 + 0], py = pos[atom * 3 + 1], pz = pos[atom * 3 + 2];
    atomicAdd(&s[0], px); atomicAdd(&s[1], py); atomicAdd(&s[2], pz);
    __syncthreads();
    float cx = s[0] / N_, cy = s[1] / N_, cz = s[2] / N_;
    float q = qbuf[atom];
    atomicAdd(&s[3], aen[atom]);
    atomicAdd(&s[4], q * (px - cx) + adip[atom * 3 + 0]);
    atomicAdd(&s[5], q * (py - cy) + adip[atom * 3 + 1]);
    atomicAdd(&s[6], q * (pz - cz) + adip[atom * 3 + 2]);
    float cacc = 0.0f;
    for (int e = t; e < EPB; e += 64) {
        int d  = dst_idx[e] + b * N_;
        int sA = src_idx[e] + b * N_;
        cacc += qbuf[sA] * qbuf[d] / (rbuf[b * EPB + e] + 1e-10f);
    }
    atomicAdd(&s[7], cacc);
    __syncthreads();
    if (t == 0) {
        out[b]              = s[3] + 0.5f * s[7] * COULOMB_SCALE_;
        out[B_ + b * 3 + 0] = s[4];
        out[B_ + b * 3 + 1] = s[5];
        out[B_ + b * 3 + 2] = s[6];
    }
}

// ---------------------------------------------------------------------------
extern "C" void kernel_launch(void* const* d_in, const int* in_sizes, int n_in,
                              void* d_out, int out_size, void* d_ws,
                              size_t ws_size, hipStream_t stream) {
    const int*   atomic_numbers = (const int*)d_in[0];
    const float* positions      = (const float*)d_in[1];
    const float* Ef             = (const float*)d_in[2];
    const int*   dst_idx        = (const int*)d_in[3];
    const int*   src_idx        = (const int*)d_in[4];
    const float* embed          = (const float*)d_in[5];
    const float* ef_kernel      = (const float*)d_in[6];
    const float* ef_bias        = (const float*)d_in[7];
    const float* mp_weights     = (const float*)d_in[8];
    const float* dense_w        = (const float*)d_in[9];
    const float* dense_b        = (const float*)d_in[10];
    const float* charge_w       = (const float*)d_in[11];
    const float* charge_b       = (const float*)d_in[12];
    const float* td_w0          = (const float*)d_in[13];
    const float* td_w1          = (const float*)d_in[14];
    const float* dip_w          = (const float*)d_in[15];
    const float* energy_w       = (const float*)d_in[16];
    const float* energy_b       = (const float*)d_in[17];
    const float* element_bias   = (const float*)d_in[18];
    float* out = (float*)d_out;

    // Workspace layout (~19 MB total)
    char* p = (char*)d_ws;
    _Float16* b0sw = (_Float16*)p; p += (size_t)A_ * 4 * 32 * 16 * sizeof(_Float16);      // 4 MB
    _Float16* b1sw = (_Float16*)p; p += (size_t)A_ * 3 * 4 * 32 * 16 * sizeof(_Float16);  // 12 MB
    float* rbuf   = (float*)p;    p += (size_t)E_ * sizeof(float);
    float* xbuf   = (float*)p;    p += (size_t)A_ * 4 * F_ * sizeof(float);
    float* ybuf   = (float*)p;    p += (size_t)A_ * 4 * F_ * sizeof(float);
    float* efp    = (float*)p;    p += (size_t)B_ * F_ * sizeof(float);
    float* qbuf   = (float*)p;    p += (size_t)A_ * sizeof(float);
    float* aenb   = (float*)p;    p += (size_t)A_ * sizeof(float);
    float* adipb  = (float*)p;    p += (size_t)A_ * 3 * sizeof(float);
    _Float16* wh  = (_Float16*)p; p += (size_t)(NITER * 4 * K_ * F_ +
                                                NITER * 3 * F_ * F_ +
                                                2 * F_ * F_) * sizeof(_Float16);
    _Float16* wh_mp    = wh;
    _Float16* wh_dense = wh + NITER * 4 * K_ * F_;
    _Float16* wh_td0   = wh_dense + NITER * 3 * F_ * F_;
    _Float16* wh_td1   = wh_td0 + F_ * F_;

    prep_kernel<<<512, 256, 0, stream>>>(mp_weights, dense_w, td_w0, td_w1, Ef,
                                         ef_kernel, ef_bias, atomic_numbers,
                                         embed, wh, efp, xbuf);
    geom_kernel<<<(A_ * 64 + 255) / 256, 256, 0, stream>>>(positions, src_idx,
                                                           b0sw, b1sw, rbuf);
    for (int it = 0; it < NITER; ++it) {
        mp_kernel<<<A_, 128, 0, stream>>>(xbuf, ybuf, b0sw, b1sw, src_idx,
                                          wh_mp + (size_t)it * 4 * K_ * F_,
                                          (it == NITER - 1) ? 1 : 0);
        update_kernel<<<A_ / 4, 32, 0, stream>>>(
            xbuf, ybuf, efp, wh_dense + (size_t)it * 3 * F_ * F_,
            dense_b + (size_t)it * 3 * F_);
    }
    heads_kernel<<<A_ / 16, 32, 0, stream>>>(
        xbuf, wh_td0, wh_td1, dip_w, charge_w, charge_b, energy_w, energy_b,
        element_bias, atomic_numbers, qbuf, aenb, adipb);
    finalize_kernel<<<B_, 64, 0, stream>>>(positions, dst_idx, src_idx, rbuf,
                                           qbuf, aenb, adipb, out);
}
